// BasicDGCNN_80616536146235
// MI455X (gfx1250) — matrix operand order repaired and checked by reading, not compiled, not run on hardware
//
#include <hip/hip_runtime.h>
#include <math.h>

#define BATCH 256
#define NPTS  16384
#define GS    208   // padded per-batch g stride (201 used)

typedef __attribute__((ext_vector_type(16))) _Float16 v16h;
typedef __attribute__((ext_vector_type(8)))  float    v8f;

// ---------------- Kernel A: per-batch coordinate first/second moments ----------------
__global__ void k_stats(const float* __restrict__ x, float* __restrict__ stats) {
  int b = blockIdx.x;
  int tid = threadIdx.x;
  const float* xb = x + (size_t)b * NPTS * 5;
  float sx=0,sy=0,sz=0,sxx=0,sxy=0,sxz=0,syy=0,syz=0,szz=0;
  for (int i = tid; i < NPTS; i += 256) {
    float px = xb[i*5+0], py = xb[i*5+1], pz = xb[i*5+2];
    sx += px; sy += py; sz += pz;
    sxx = fmaf(px,px,sxx); sxy = fmaf(px,py,sxy); sxz = fmaf(px,pz,sxz);
    syy = fmaf(py,py,syy); syz = fmaf(py,pz,syz); szz = fmaf(pz,pz,szz);
  }
  __shared__ float red[256];
  __shared__ float out9[9];
  float vals[9] = {sx,sy,sz,sxx,sxy,sxz,syy,syz,szz};
  for (int j = 0; j < 9; ++j) {
    red[tid] = vals[j]; __syncthreads();
    for (int off = 128; off > 0; off >>= 1) {
      if (tid < off) red[tid] += red[tid+off];
      __syncthreads();
    }
    if (tid == 0) out9[j] = red[0];
    __syncthreads();
  }
  if (tid < 9) stats[b*12+tid] = out9[tid];
}

// ---------------- Kernel B: analytic symmetric 3x3 eigendecomposition ----------------
__device__ inline void cross3(const float* a, const float* bb, float* c) {
  c[0] = a[1]*bb[2] - a[2]*bb[1];
  c[1] = a[2]*bb[0] - a[0]*bb[2];
  c[2] = a[0]*bb[1] - a[1]*bb[0];
}

__device__ inline void eigvec(float a00,float a01,float a02,float a11,float a12,float a22,
                              float lam, float* v) {
  float r0[3] = {a00-lam, a01, a02};
  float r1[3] = {a01, a11-lam, a12};
  float r2[3] = {a02, a12, a22-lam};
  float c0[3], c1[3], c2[3];
  cross3(r0,r1,c0); cross3(r0,r2,c1); cross3(r1,r2,c2);
  float n0 = c0[0]*c0[0]+c0[1]*c0[1]+c0[2]*c0[2];
  float n1 = c1[0]*c1[0]+c1[1]*c1[1]+c1[2]*c1[2];
  float n2 = c2[0]*c2[0]+c2[1]*c2[1]+c2[2]*c2[2];
  const float* cb = c0; float nb = n0;
  if (n1 > nb) { cb = c1; nb = n1; }
  if (n2 > nb) { cb = c2; nb = n2; }
  if (nb > 1e-30f) {
    float inv = rsqrtf(nb);
    v[0] = cb[0]*inv; v[1] = cb[1]*inv; v[2] = cb[2]*inv;
  } else {
    v[0] = 1.0f; v[1] = 0.0f; v[2] = 0.0f;
  }
}

__global__ void k_eigen(const float* __restrict__ stats, float* __restrict__ evc,
                        float* __restrict__ g) {
  int b = blockIdx.x * blockDim.x + threadIdx.x;
  if (b >= BATCH) return;
  const float invN = 1.0f / NPTS;
  float mx = stats[b*12+0]*invN, my = stats[b*12+1]*invN, mz = stats[b*12+2]*invN;
  float a00 = stats[b*12+3]*invN - mx*mx;
  float a01 = stats[b*12+4]*invN - mx*my;
  float a02 = stats[b*12+5]*invN - mx*mz;
  float a11 = stats[b*12+6]*invN - my*my;
  float a12 = stats[b*12+7]*invN - my*mz;
  float a22 = stats[b*12+8]*invN - mz*mz;

  float p1 = a01*a01 + a02*a02 + a12*a12;
  float q  = (a00+a11+a22) * (1.0f/3.0f);
  float e1, e2, e3;
  if (p1 < 1e-30f) {
    e1 = fmaxf(a00, fmaxf(a11, a22));
    e3 = fminf(a00, fminf(a11, a22));
    e2 = a00 + a11 + a22 - e1 - e3;
  } else {
    float p2 = (a00-q)*(a00-q) + (a11-q)*(a11-q) + (a22-q)*(a22-q) + 2.0f*p1;
    float p  = sqrtf(p2 * (1.0f/6.0f));
    float ip = 1.0f / p;
    float b00=(a00-q)*ip, b01=a01*ip, b02=a02*ip;
    float b11=(a11-q)*ip, b12=a12*ip, b22=(a22-q)*ip;
    float detB = b00*(b11*b22 - b12*b12) - b01*(b01*b22 - b12*b02) + b02*(b01*b12 - b11*b02);
    float r = fminf(1.0f, fmaxf(-1.0f, 0.5f*detB));
    float phi = acosf(r) * (1.0f/3.0f);
    e1 = q + 2.0f*p*cosf(phi);
    e3 = q + 2.0f*p*cosf(phi + 2.0943951023931953f);
    e2 = 3.0f*q - e1 - e3;
  }
  float v1[3], v2[3], v3[3];
  eigvec(a00,a01,a02,a11,a12,a22, e1, v1);
  eigvec(a00,a01,a02,a11,a12,a22, e3, v3);
  cross3(v1, v3, v2);
  float n2 = v2[0]*v2[0]+v2[1]*v2[1]+v2[2]*v2[2];
  float inv2 = (n2 > 1e-30f) ? rsqrtf(n2) : 0.0f;
  v2[0]*=inv2; v2[1]*=inv2; v2[2]*=inv2;

  float* E = evc + b*12;
  E[0]=v1[0]; E[1]=v1[1]; E[2]=v1[2];
  E[3]=v2[0]; E[4]=v2[1]; E[5]=v2[2];
  E[6]=v3[0]; E[7]=v3[1]; E[8]=v3[2];
  E[9]=mx; E[10]=my; E[11]=mz;

  float inv_tr = 1.0f / (e1 + e2 + e3 + 1e-8f);
  float* gb = g + (size_t)b * GS;
  gb[192] = e1*inv_tr; gb[193] = e2*inv_tr; gb[194] = e3*inv_tr;
  gb[198] = mx; gb[199] = my; gb[200] = mz;
}

// ---------------- Kernel C: extents = max-min of projections onto eigenvectors ----------------
__global__ void k_extents(const float* __restrict__ x, const float* __restrict__ evc,
                          float* __restrict__ g) {
  int b = blockIdx.x, tid = threadIdx.x;
  const float* E = evc + b*12;
  float e00=E[0],e01=E[1],e02=E[2], e10=E[3],e11=E[4],e12=E[5], e20=E[6],e21=E[7],e22=E[8];
  float cx=E[9], cy=E[10], cz=E[11];
  const float* xb = x + (size_t)b * NPTS * 5;
  float mn0=1e30f, mn1=1e30f, mn2=1e30f, mx0=-1e30f, mx1=-1e30f, mx2=-1e30f;
  for (int i = tid; i < NPTS; i += 256) {
    float px = xb[i*5+0]-cx, py = xb[i*5+1]-cy, pz = xb[i*5+2]-cz;
    float p0 = px*e00 + py*e01 + pz*e02;
    float p1 = px*e10 + py*e11 + pz*e12;
    float p2 = px*e20 + py*e21 + pz*e22;
    mn0 = fminf(mn0,p0); mx0 = fmaxf(mx0,p0);
    mn1 = fminf(mn1,p1); mx1 = fmaxf(mx1,p1);
    mn2 = fminf(mn2,p2); mx2 = fmaxf(mx2,p2);
  }
  __shared__ float red[256];
  __shared__ float out6[6];
  float vals[6] = {mn0,mn1,mn2,mx0,mx1,mx2};
  for (int j = 0; j < 6; ++j) {
    red[tid] = vals[j]; __syncthreads();
    for (int off = 128; off > 0; off >>= 1) {
      if (tid < off) red[tid] = (j < 3) ? fminf(red[tid], red[tid+off])
                                        : fmaxf(red[tid], red[tid+off]);
      __syncthreads();
    }
    if (tid == 0) out6[j] = red[0];
    __syncthreads();
  }
  if (tid < 3) g[(size_t)b*GS + 195 + tid] = out6[3+tid] - out6[tid];
}

// ---------------- Kernel D: per-point MLP (WMMA f16 16x16x32) + max/mean/std pooling ----------------
__global__ void __launch_bounds__(256) k_pointmlp(const float* __restrict__ x,
                           const float* __restrict__ W1, const float* __restrict__ b1,
                           const float* __restrict__ W2, const float* __restrict__ b2,
                           float* __restrict__ g) {
  const int b = blockIdx.x;
  const int tid = threadIdx.x;
  const int lane = tid & 31;
  const int w = tid >> 5;            // wave in block, 0..7
  const int col = lane & 15;
  const int hh = (lane >> 4) & 1;    // lane half
  const int abase = hh ? 8 : 0;      // A-tile K base for this half (16-bit A layout)
  const int kbase = hh ? 16 : 0;     // unused; B tiles use abase, same k order as A

  // Per-lane W1/b1 columns matching this lane's A-operand K slots.
  float w1a[16], w1b[16], b1r[16];
#pragma unroll
  for (int e = 0; e < 16; ++e) {
    int k = (e < 8) ? (abase + e) : (16 + abase + (e - 8));
    w1a[e] = W1[k];
    w1b[e] = W1[32 + k];
    b1r[e] = b1[k];
  }
  // Preload W2 (32x64) as 4 B-matrix register tiles (N tiles of 16).
  v16h bt[4];
#pragma unroll
  for (int t = 0; t < 4; ++t) {
#pragma unroll
    for (int e = 0; e < 16; ++e) {
      bt[t][e] = (_Float16)W2[((e < 8) ? (abase + e) : (16 + abase + (e - 8))) * 64 + t * 16 + col];
    }
  }
  // Bias folded into the WMMA accumulator: every C slot of this lane in tile t
  // is output column N = 16t+col, so a b2-filled C gives D = A*B + b2 for free.
  v8f cbias[4];
#pragma unroll
  for (int t = 0; t < 4; ++t) {
    float bv = b2[t * 16 + col];
#pragma unroll
    for (int r = 0; r < 8; ++r) cbias[t][r] = bv;
  }

  float amax[4] = {0,0,0,0}, asum[4] = {0,0,0,0}, asq[4] = {0,0,0,0};
  const float* xb = x + (size_t)b * NPTS * 5;

  for (int it = 0; it < NPTS / (16 * 8); ++it) {
    int p = (it * 8 + w) * 16 + col;           // this lane's point row (M = col)
    float s0 = xb[p * 5 + 3];
    float s1 = xb[p * 5 + 4];
    // Layer 1 in fp32, written directly into A-operand f16 layout.
    v16h a;
#pragma unroll
    for (int e = 0; e < 16; ++e) {
      float v = fmaf(s1, w1b[e], fmaf(s0, w1a[e], b1r[e]));
      a[e] = (_Float16)fmaxf(v, 0.0f);
    }
    // Layer 2: 4 WMMAs (16 points x 32 hidden x 64 out), bias via C, fused relu/pool.
#pragma unroll
    for (int t = 0; t < 4; ++t) {
      v8f c = __builtin_amdgcn_wmma_f32_16x16x32_f16(false, a, false, bt[t],
                                                     (short)0, cbias[t], false, false);
#pragma unroll
      for (int r = 0; r < 8; ++r) {
        float v = fmaxf(c[r], 0.0f);
        amax[t] = fmaxf(amax[t], v);
        asum[t] += v;
        asq[t]  = fmaf(v, v, asq[t]);
      }
    }
  }

  // Block reduction: 16 slots (8 waves x 2 lane-halves) per feature.
  __shared__ float smax[16*64], ssum[16*64], ssq[16*64];
  int slot = w * 2 + hh;
#pragma unroll
  for (int t = 0; t < 4; ++t) {
    int n = t * 16 + col;
    smax[slot*64 + n] = amax[t];
    ssum[slot*64 + n] = asum[t];
    ssq [slot*64 + n] = asq[t];
  }
  __syncthreads();
  if (tid < 64) {
    float mx = smax[tid], sm = ssum[tid], sq = ssq[tid];
    for (int s = 1; s < 16; ++s) {
      mx = fmaxf(mx, smax[s*64 + tid]);
      sm += ssum[s*64 + tid];
      sq += ssq [s*64 + tid];
    }
    const float invN = 1.0f / NPTS;
    float mean = sm * invN;
    float var  = (sq - sm * mean) / (float)(NPTS - 1);   // ddof=1
    float* gb = g + (size_t)b * GS;
    gb[tid]        = mx;
    gb[64 + tid]   = mean;
    gb[128 + tid]  = sqrtf(fmaxf(var, 0.0f));
  }
}

// ---------------- Kernel E: head MLP 201 -> 256 -> 128 -> 256 ----------------
__global__ void k_head(const float* __restrict__ g,
                       const float* __restrict__ W3, const float* __restrict__ b3,
                       const float* __restrict__ W4, const float* __restrict__ b4,
                       const float* __restrict__ W5, const float* __restrict__ b5,
                       float* __restrict__ out) {
  int b = blockIdx.x, tid = threadIdx.x;
  __shared__ float sg[201];
  __shared__ float sh1[256];
  __shared__ float sh2[128];
  if (tid < 201) sg[tid] = g[(size_t)b * GS + tid];
  __syncthreads();
  float acc = b3[tid];
  for (int k = 0; k < 201; ++k) acc = fmaf(sg[k], W3[k*256 + tid], acc);
  sh1[tid] = fmaxf(acc, 0.0f);
  __syncthreads();
  if (tid < 128) {
    float a2 = b4[tid];
    for (int k = 0; k < 256; ++k) a2 = fmaf(sh1[k], W4[k*128 + tid], a2);
    sh2[tid] = fmaxf(a2, 0.0f);
  }
  __syncthreads();
  float a3 = b5[tid];
  for (int k = 0; k < 128; ++k) a3 = fmaf(sh2[k], W5[k*256 + tid], a3);
  out[(size_t)b * 256 + tid] = a3;
}

extern "C" void kernel_launch(void* const* d_in, const int* in_sizes, int n_in,
                              void* d_out, int out_size, void* d_ws, size_t ws_size,
                              hipStream_t stream) {
  (void)in_sizes; (void)n_in; (void)out_size; (void)ws_size;
  const float* x  = (const float*)d_in[0];
  const float* W1 = (const float*)d_in[1];
  const float* b1 = (const float*)d_in[2];
  const float* W2 = (const float*)d_in[3];
  const float* b2 = (const float*)d_in[4];
  const float* W3 = (const float*)d_in[5];
  const float* b3 = (const float*)d_in[6];
  const float* W4 = (const float*)d_in[7];
  const float* b4 = (const float*)d_in[8];
  const float* W5 = (const float*)d_in[9];
  const float* b5 = (const float*)d_in[10];
  float* out = (float*)d_out;

  float* ws    = (float*)d_ws;
  float* stats = ws;                 // BATCH*12 floats
  float* evc   = ws + BATCH * 12;    // BATCH*12 floats
  float* g     = ws + BATCH * 24;    // BATCH*GS floats

  k_stats   <<<BATCH, 256, 0, stream>>>(x, stats);
  k_eigen   <<<1,     256, 0, stream>>>(stats, evc, g);
  k_extents <<<BATCH, 256, 0, stream>>>(x, evc, g);
  k_pointmlp<<<BATCH, 256, 0, stream>>>(x, W1, b1, W2, b2, g);
  k_head    <<<BATCH, 256, 0, stream>>>(g, W3, b3, W4, b4, W5, b5, out);
}
